// DetModel_90366111908255
// MI455X (gfx1250) — matrix operand order repaired and checked by reading, not compiled, not run on hardware
//
#include <hip/hip_runtime.h>
#include <math.h>

typedef __attribute__((ext_vector_type(16))) _Float16 v16h;
typedef __attribute__((ext_vector_type(8)))  _Float16 v8h;
typedef __attribute__((ext_vector_type(4)))  _Float16 v4h;
typedef __attribute__((ext_vector_type(8)))  float    v8f;

union AV { v16h v; v8h h[2]; };

#define B_  8
#define H_  256
#define W_  256
#define C_  128
#define NBOX 2048
#define MT  4            // M tiles (of 16 pixels) per block -> 64 pixels/block
#define PCOLS (MT * 16 + 2)   // 66 patch columns

// ---------------------------------------------------------------------------
// f32 -> f16 elementwise convert (vectorized by 4)
// ---------------------------------------------------------------------------
__global__ __launch_bounds__(256) void k_cvt_f16(const float* __restrict__ src,
                                                 _Float16* __restrict__ dst, int n4) {
  int i = blockIdx.x * 256 + threadIdx.x;
  if (i >= n4) return;
  const float4 f = ((const float4*)src)[i];
  v4h o; o.x = (_Float16)f.x; o.y = (_Float16)f.y; o.z = (_Float16)f.z; o.w = (_Float16)f.w;
  *(v4h*)(dst + (size_t)i * 4) = o;
}

// ---------------------------------------------------------------------------
// Repack conv weights (3,3,128,128 HWIO f32) into per-lane WMMA B-fragment
// order: wf[((kc*8+nt)*32+lane)*16 + e] = f16(w[(kc*32 + (lane>=16?16:0)+e)*128 + nt*16+lane%16])
// ---------------------------------------------------------------------------
__global__ __launch_bounds__(256) void k_repack_w(const float* __restrict__ w,
                                                  _Float16* __restrict__ wf) {
  int t = blockIdx.x * 256 + threadIdx.x;           // 36*8*32 = 9216
  if (t >= 36 * 8 * 32) return;
  int lane = t & 31;
  int nt   = (t >> 5) & 7;
  int kc   = t >> 8;
  int kbase = kc * 32 + ((lane >= 16) ? 16 : 0);
  int n     = nt * 16 + (lane & 15);
  AV frag;
#pragma unroll
  for (int e = 0; e < 16; ++e) frag.h[e >> 3][e & 7] = (_Float16)w[(size_t)(kbase + e) * C_ + n];
  *(v16h*)(wf + (size_t)t * 16) = frag.v;
}

// ---------------------------------------------------------------------------
// 3x3 SAME conv, C=128 -> 128, fused bias+ReLU, f16 in/out, f32 WMMA accum.
// Block: 256 threads = 8 waves; 64 output pixels (along W) x 128 channels.
// Each wave: one 16-channel N-tile, 4 M-tiles (4 f32 accumulators), so each
// 32B B-fragment load feeds 4 WMMAs (4x less L2 weight traffic than 1 tile).
// ---------------------------------------------------------------------------
__global__ __launch_bounds__(256) void k_conv3x3_relu(const _Float16* __restrict__ in,
                                                      const _Float16* __restrict__ wf,
                                                      const float* __restrict__ bias,
                                                      _Float16* __restrict__ out) {
  __shared__ _Float16 patch[3 * PCOLS * C_];        // 3*66*128*2 = 50688 B
  const int tid = threadIdx.x;
  const int bid = blockIdx.x;                        // B*H*(W/64) = 8192
  const int wt = bid & 3;
  const int h  = (bid >> 2) & 255;
  const int b  = bid >> 10;
  const int w0 = wt << 6;

  // cooperative zero-padded patch load: 3 rows x 66 cols x 128ch, 8 halfs/iter
  for (int ch = tid; ch < 3 * PCOLS * 16; ch += 256) {
    int row = ch / (PCOLS * 16);
    int r   = ch % (PCOLS * 16);
    int col = r >> 4;
    int c8  = (r & 15) << 3;
    int hin = h + row - 1;
    int win = w0 + col - 1;
    v8h v = {};
    if (hin >= 0 && hin < H_ && win >= 0 && win < W_)
      v = *(const v8h*)(in + (((size_t)b * H_ + hin) * W_ + win) * C_ + c8);
    *(v8h*)(patch + (row * PCOLS + col) * C_ + c8) = v;
  }
  __syncthreads();

  const int lane = tid & 31;
  const int nt   = tid >> 5;                         // wave = output-channel tile
  const int m    = lane & 15;                        // pixel within M tile
  const int kb8  = (lane >= 16) ? 16 : 0;

  v8f acc[MT] = {};
#pragma unroll 4
  for (int kc = 0; kc < 36; ++kc) {
    const int ky   = kc / 12;
    const int kx   = (kc % 12) >> 2;
    const int cinb = (kc & 3) << 5;
    // B-fragment: one contiguous 32B coalesced load from fragment-ordered weights
    const v16h bf = *(const v16h*)(wf + ((size_t)((kc << 3) + nt) * 32 + lane) * 16);
#pragma unroll
    for (int mt = 0; mt < MT; ++mt) {
      // A-fragment from LDS: two aligned b128 runs per lane (16-bit A 16x32 layout)
      const _Float16* ap = patch + ((ky * PCOLS) + mt * 16 + m + kx) * C_ + cinb + kb8;
      AV a;
      a.h[0] = *(const v8h*)ap;
      a.h[1] = *(const v8h*)(ap + 8);
      acc[mt] = __builtin_amdgcn_wmma_f32_16x16x32_f16(false, a.v, false, bf,
                                                       (short)0, acc[mt], false, false);
    }
  }

  const int n    = (nt << 4) + (lane & 15);
  const float bb = bias[n];
  const int mrow = (lane >= 16) ? 8 : 0;
#pragma unroll
  for (int mt = 0; mt < MT; ++mt) {
    const size_t ob = (((size_t)b * H_ + h) * W_ + w0 + mt * 16) * C_ + n;
#pragma unroll
    for (int r = 0; r < 8; ++r) {
      float v = acc[mt][r] + bb;
      v = v > 0.f ? v : 0.f;
      out[ob + (size_t)(mrow + r) * C_] = (_Float16)v;
    }
  }
}

// ---------------------------------------------------------------------------
// Fused 1x1 heads + class selection: only the channels picked by cls[b] are
// computed. One thread per pixel.
// ---------------------------------------------------------------------------
__global__ __launch_bounds__(256) void k_head(const _Float16* __restrict__ xreg,
                                              const _Float16* __restrict__ xsz,
                                              const float* __restrict__ ofs_w,
                                              const float* __restrict__ ofs_b,
                                              const float* __restrict__ wt_w,
                                              const float* __restrict__ wt_b,
                                              const float* __restrict__ size_w,
                                              const float* __restrict__ size_b,
                                              const int* __restrict__ cls,
                                              float* __restrict__ o_ofs,
                                              float* __restrict__ o_sz,
                                              float* __restrict__ o_wt) {
  const int p = blockIdx.x * 256 + threadIdx.x;      // B*H*W = 524288
  if (p >= B_ * H_ * W_) return;
  const int b = p >> 16;
  const int c = cls[b];
  const _Float16* xr = xreg + (size_t)p * C_;
  const _Float16* xs = xsz  + (size_t)p * C_;
  float d0 = 0.f, d1 = 0.f, dw = 0.f, s0 = 0.f, s1 = 0.f;
#pragma unroll 8
  for (int ch = 0; ch < C_; ++ch) {
    const float v = (float)xr[ch];
    d0 += v * ofs_w[ch * 6 + 2 * c];
    d1 += v * ofs_w[ch * 6 + 2 * c + 1];
    dw += v * wt_w[ch * 3 + c];
    const float u = (float)xs[ch];
    s0 += u * size_w[ch * 6 + 2 * c];
    s1 += u * size_w[ch * 6 + 2 * c + 1];
  }
  o_ofs[(size_t)p * 2 + 0] = d0 + ofs_b[2 * c];
  o_ofs[(size_t)p * 2 + 1] = d1 + ofs_b[2 * c + 1];
  o_sz [(size_t)p * 2 + 0] = s0 + size_b[2 * c];
  o_sz [(size_t)p * 2 + 1] = s1 + size_b[2 * c + 1];
  o_wt [p] = dw + wt_b[c];
}

// ---------------------------------------------------------------------------
// cls branch: stride-3 VALID conv (16x16x256 -> 5x5x128) + mean + 2 FCs + softmax
// One block per batch.
// ---------------------------------------------------------------------------
__global__ __launch_bounds__(256) void k_cls(const float* __restrict__ feat,
                                             const float* __restrict__ cw,
                                             const float* __restrict__ cb,
                                             const float* __restrict__ fw1,
                                             const float* __restrict__ fb1,
                                             const float* __restrict__ fw2,
                                             const float* __restrict__ fb2,
                                             const float* __restrict__ ow,
                                             const float* __restrict__ ob,
                                             float* __restrict__ pred) {
  __shared__ float c0[128], c1[128], c2[128];
  const int b = blockIdx.x, tid = threadIdx.x;
  if (tid < 128) c0[tid] = 0.f;
  __syncthreads();
  for (int task = tid; task < 128 * 25; task += 256) {
    const int o = task / 25, pos = task % 25;
    const int oy = pos / 5, ox = pos % 5;
    float s = cb[o];
    for (int ky = 0; ky < 3; ++ky)
      for (int kx = 0; kx < 3; ++kx) {
        const float* fp = feat + (((size_t)b * 16 + oy * 3 + ky) * 16 + ox * 3 + kx) * 256;
        const float* wp = cw + (size_t)((ky * 3 + kx) * 256) * 128 + o;
        for (int ci = 0; ci < 256; ++ci) s += fp[ci] * wp[(size_t)ci * 128];
      }
    s = s > 0.f ? s : 0.f;
    atomicAdd(&c0[o], s * 0.04f);                    // 1/25 mean
  }
  __syncthreads();
  if (tid < 128) {
    float s = fb1[tid];
    for (int ci = 0; ci < 128; ++ci) s += c0[ci] * fw1[ci * 128 + tid];
    c1[tid] = s > 0.f ? s : 0.f;
  }
  __syncthreads();
  if (tid < 128) {
    float s = fb2[tid];
    for (int ci = 0; ci < 128; ++ci) s += c1[ci] * fw2[ci * 128 + tid];
    c2[tid] = s > 0.f ? s : 0.f;
  }
  __syncthreads();
  if (tid == 0) {
    float l[3], mx = -1e30f;
    for (int j = 0; j < 3; ++j) {
      float s = ob[j];
      for (int ci = 0; ci < 128; ++ci) s += c2[ci] * ow[ci * 3 + j];
      l[j] = s; mx = s > mx ? s : mx;
    }
    float den = 0.f;
    for (int j = 0; j < 3; ++j) { l[j] = __expf(l[j] - mx); den += l[j]; }
    for (int j = 0; j < 3; ++j) pred[b * 3 + j] = l[j] / den;
  }
}

// ---------------------------------------------------------------------------
// NMS: one block per batch. Bitonic sort 2048 scores (desc), sequential greedy
// suppression, MAX_OUT cap, masked writeback in sorted order.
// ---------------------------------------------------------------------------
__global__ __launch_bounds__(256) void k_nms(const float* __restrict__ boxes,
                                             const float* __restrict__ scores,
                                             float* __restrict__ o_boxes,
                                             float* __restrict__ o_scores) {
  __shared__ float ls[NBOX];
  __shared__ int   lidx[NBOX];
  __shared__ float bs[NBOX * 4];
  __shared__ int   keep[NBOX];
  const int b = blockIdx.x, tid = threadIdx.x;

  for (int i = tid; i < NBOX; i += 256) { ls[i] = scores[b * NBOX + i]; lidx[i] = i; }
  __syncthreads();

  for (int k = 2; k <= NBOX; k <<= 1)
    for (int j = k >> 1; j > 0; j >>= 1) {
      for (int i = tid; i < NBOX; i += 256) {
        const int ixj = i ^ j;
        if (ixj > i) {
          const bool desc = ((i & k) == 0);
          const float a = ls[i], bb = ls[ixj];
          if ((a < bb) == desc) {
            ls[i] = bb; ls[ixj] = a;
            const int t = lidx[i]; lidx[i] = lidx[ixj]; lidx[ixj] = t;
          }
        }
      }
      __syncthreads();
    }

  for (int i = tid; i < NBOX; i += 256) {
    const int oi = lidx[i];
    bs[i * 4 + 0] = boxes[(b * NBOX + oi) * 4 + 0];
    bs[i * 4 + 1] = boxes[(b * NBOX + oi) * 4 + 1];
    bs[i * 4 + 2] = boxes[(b * NBOX + oi) * 4 + 2];
    bs[i * 4 + 3] = boxes[(b * NBOX + oi) * 4 + 3];
    keep[i] = (ls[i] >= 0.05f) ? 1 : 0;
  }

  for (int i = 0; i < NBOX; ++i) {
    __syncthreads();
    if (!keep[i]) continue;
    const float y1 = bs[i * 4 + 0], x1 = bs[i * 4 + 1], y2 = bs[i * 4 + 2], x2 = bs[i * 4 + 3];
    const float ai = (y2 - y1) * (x2 - x1);
    for (int j = i + 1 + tid; j < NBOX; j += 256) {
      if (!keep[j]) continue;
      const float yy1 = fmaxf(y1, bs[j * 4 + 0]);
      const float xx1 = fmaxf(x1, bs[j * 4 + 1]);
      const float yy2 = fminf(y2, bs[j * 4 + 2]);
      const float xx2 = fminf(x2, bs[j * 4 + 3]);
      const float inter = fmaxf(yy2 - yy1, 0.f) * fmaxf(xx2 - xx1, 0.f);
      const float aj = (bs[j * 4 + 2] - bs[j * 4 + 0]) * (bs[j * 4 + 3] - bs[j * 4 + 1]);
      const float iou = inter / (ai + aj - inter + 1e-8f);
      if (iou > 0.5f) keep[j] = 0;
    }
  }
  __syncthreads();
  if (tid == 0) {
    int cnt = 0;
    for (int i = 0; i < NBOX; ++i)
      if (keep[i]) { ++cnt; if (cnt > 2000) keep[i] = 0; }
  }
  __syncthreads();
  for (int i = tid; i < NBOX; i += 256) {
    const int k = keep[i];
    o_boxes[(b * NBOX + i) * 4 + 0] = k ? bs[i * 4 + 0] : 0.f;
    o_boxes[(b * NBOX + i) * 4 + 1] = k ? bs[i * 4 + 1] : 0.f;
    o_boxes[(b * NBOX + i) * 4 + 2] = k ? bs[i * 4 + 2] : 0.f;
    o_boxes[(b * NBOX + i) * 4 + 3] = k ? bs[i * 4 + 3] : 0.f;
    o_scores[b * NBOX + i] = k ? ls[i] : 0.f;
  }
}

// ---------------------------------------------------------------------------
extern "C" void kernel_launch(void* const* d_in, const int* in_sizes, int n_in,
                              void* d_out, int out_size, void* d_ws, size_t ws_size,
                              hipStream_t stream) {
  const float* feat_reg = (const float*)d_in[0];
  const float* feat_cls = (const float*)d_in[1];
  const int*   cls      = (const int*)d_in[2];
  const float* boxes    = (const float*)d_in[3];
  const float* scores   = (const float*)d_in[4];
  const float* reg_w1 = (const float*)d_in[5];  const float* reg_b1 = (const float*)d_in[6];
  const float* reg_w2 = (const float*)d_in[7];  const float* reg_b2 = (const float*)d_in[8];
  const float* sz_w1  = (const float*)d_in[9];  const float* sz_b1  = (const float*)d_in[10];
  const float* sz_w2  = (const float*)d_in[11]; const float* sz_b2  = (const float*)d_in[12];
  const float* ofs_w  = (const float*)d_in[13]; const float* ofs_b  = (const float*)d_in[14];
  const float* wt_w   = (const float*)d_in[15]; const float* wt_b   = (const float*)d_in[16];
  const float* size_w = (const float*)d_in[17]; const float* size_b = (const float*)d_in[18];
  const float* cls_cw = (const float*)d_in[19]; const float* cls_cb = (const float*)d_in[20];
  const float* cls_fw1= (const float*)d_in[21]; const float* cls_fb1= (const float*)d_in[22];
  const float* cls_fw2= (const float*)d_in[23]; const float* cls_fb2= (const float*)d_in[24];
  const float* cls_ow = (const float*)d_in[25]; const float* cls_ob = (const float*)d_in[26];

  float* out = (float*)d_out;
  float* o_ofs = out;                 // (8,256,256,2)
  float* o_sz  = out + 1048576;       // (8,256,256,2)
  float* o_wt  = out + 2097152;       // (8,256,256,1)
  float* o_cls = out + 2621440;       // (8,3)
  float* o_nb  = out + 2621464;       // (8,2048,4)
  float* o_ns  = out + 2687000;       // (8,2048)

  // workspace carve (bytes)
  const size_t FEAT16 = (size_t)B_ * H_ * W_ * C_ * sizeof(_Float16);   // 128 MiB
  const size_t WF     = (size_t)36 * 8 * 32 * 16 * sizeof(_Float16);    // 288 KiB
  char* ws = (char*)d_ws;
  _Float16* featf16 = (_Float16*)(ws);
  _Float16* tmp16   = (_Float16*)(ws + FEAT16);
  _Float16* xreg16  = (_Float16*)(ws + 2 * FEAT16);
  _Float16* xsz16   = featf16;        // reuse featf16 region after last read
  _Float16* wf1 = (_Float16*)(ws + 3 * FEAT16);
  _Float16* wf2 = (_Float16*)(ws + 3 * FEAT16 + WF);
  _Float16* wf3 = (_Float16*)(ws + 3 * FEAT16 + 2 * WF);
  _Float16* wf4 = (_Float16*)(ws + 3 * FEAT16 + 3 * WF);

  const int nfeat4 = B_ * H_ * W_ * C_ / 4;                             // 16777216
  k_cvt_f16<<<nfeat4 / 256, 256, 0, stream>>>(feat_reg, featf16, nfeat4);

  k_repack_w<<<36, 256, 0, stream>>>(reg_w1, wf1);
  k_repack_w<<<36, 256, 0, stream>>>(reg_w2, wf2);
  k_repack_w<<<36, 256, 0, stream>>>(sz_w1,  wf3);
  k_repack_w<<<36, 256, 0, stream>>>(sz_w2,  wf4);

  const int convGrid = B_ * H_ * (W_ / (MT * 16));                      // 8192
  k_conv3x3_relu<<<convGrid, 256, 0, stream>>>(featf16, wf1, reg_b1, tmp16);
  k_conv3x3_relu<<<convGrid, 256, 0, stream>>>(tmp16,   wf2, reg_b2, xreg16);
  k_conv3x3_relu<<<convGrid, 256, 0, stream>>>(featf16, wf3, sz_b1,  tmp16);
  k_conv3x3_relu<<<convGrid, 256, 0, stream>>>(tmp16,   wf4, sz_b2,  xsz16);

  k_head<<<(B_ * H_ * W_) / 256, 256, 0, stream>>>(xreg16, xsz16,
      ofs_w, ofs_b, wt_w, wt_b, size_w, size_b, cls, o_ofs, o_sz, o_wt);

  k_cls<<<B_, 256, 0, stream>>>(feat_cls, cls_cw, cls_cb, cls_fw1, cls_fb1,
                                cls_fw2, cls_fb2, cls_ow, cls_ob, o_cls);

  k_nms<<<B_, 256, 0, stream>>>(boxes, scores, o_nb, o_ns);
}
